// self_attn2_16088947491440
// MI455X (gfx1250) — matrix operand order repaired and checked by reading, not compiled, not run on hardware
//
#include <hip/hip_runtime.h>

#define BB   4
#define CC   256
#define C8V  32
#define NNV  4096
#define EPSV 1e-5f

typedef __attribute__((ext_vector_type(16))) __bf16 v16bf;
typedef __attribute__((ext_vector_type(2)))  __bf16 v2bf;
typedef __attribute__((ext_vector_type(8)))  float  v8f;

union Mat16 { v16bf v; unsigned int u[8]; };

// Native __bf16 conversions: let the backend pick v_cvt_pk_bf16_f32 / hw cvt.
__device__ __forceinline__ unsigned short f2bf(float f) {
    __bf16 h = (__bf16)f;
    return __builtin_bit_cast(unsigned short, h);
}
__device__ __forceinline__ unsigned int packbf(float a, float b) {
    v2bf r = { (__bf16)a, (__bf16)b };
    return __builtin_bit_cast(unsigned int, r);
}
__device__ __forceinline__ v8f wmma_bf16(const Mat16& a, const Mat16& b, v8f c) {
    return __builtin_amdgcn_wmma_f32_16x16x32_bf16(false, a.v, false, b.v,
                                                   (short)0, c, false, false);
}

// ---------------------------------------------------------------------------
// Kernel 0: x [B][C][N] f32  ->  xT [B][N][C] bf16  (32x32 LDS tile transpose)
// ---------------------------------------------------------------------------
__global__ void __launch_bounds__(256) xpose_kernel(const float* __restrict__ x,
                                                    unsigned short* __restrict__ xT) {
    __shared__ float t[32][33];
    const int b  = blockIdx.z;
    const int c0 = blockIdx.y * 32;
    const int n0 = blockIdx.x * 32;
    const int tx = threadIdx.x, ty = threadIdx.y;
    #pragma unroll
    for (int cc = ty; cc < 32; cc += 8)
        t[cc][tx] = x[((b * CC + c0 + cc) * NNV) + n0 + tx];
    __syncthreads();
    #pragma unroll
    for (int nn = ty; nn < 32; nn += 8)
        xT[((b * NNV) + n0 + nn) * CC + c0 + tx] = f2bf(t[tx][nn]);
}

// ---------------------------------------------------------------------------
// Kernel 1: fused 1x1-conv + BN for q/k/v via WMMA.
// grid = (20 outtiles, N/16, B); 1 wave/block.
//   outtile 0..1 -> q channels, 2..3 -> k channels, 4..19 -> v channels
// ---------------------------------------------------------------------------
__global__ void __launch_bounds__(32) proj_kernel(
        const unsigned short* __restrict__ xT,
        const float* __restrict__ wq, const float* __restrict__ qg,
        const float* __restrict__ qbb, const float* __restrict__ qm,
        const float* __restrict__ qv,
        const float* __restrict__ wk, const float* __restrict__ kg,
        const float* __restrict__ kbb, const float* __restrict__ km,
        const float* __restrict__ kv,
        const float* __restrict__ wv, const float* __restrict__ vg,
        const float* __restrict__ vbb, const float* __restrict__ vm,
        const float* __restrict__ vv,
        unsigned short* __restrict__ qbuf, unsigned short* __restrict__ kbuf,
        unsigned short* __restrict__ vbuf) {
    const int lane = threadIdx.x;
    const int m  = lane & 15;
    const int hi = lane >> 4;
    const int ot = blockIdx.x;
    const int i0 = blockIdx.y * 16;
    const int b  = blockIdx.z;

    const float *w, *g, *bt, *mu, *vr;
    int omat, kind;
    if (ot < 2)      { w = wq; g = qg; bt = qbb; mu = qm; vr = qv; omat = ot * 16;      kind = 0; }
    else if (ot < 4) { w = wk; g = kg; bt = kbb; mu = km; vr = kv; omat = (ot-2) * 16;  kind = 1; }
    else             { w = wv; g = vg; bt = vbb; mu = vm; vr = vv; omat = (ot-4) * 16;  kind = 2; }
    const int o = omat + m;   // output channel (D column) for this lane

    v8f acc = {};
    #pragma unroll
    for (int c0 = 0; c0 < CC; c0 += 32) {
        Mat16 a, bm;
        #pragma unroll
        for (int r = 0; r < 8; ++r) {
            const int kkA = ((r & 3) * 2) + ((r >> 2) * 16) + 8 * hi;   // A layout
            a.u[r] = *(const unsigned int*)&xT[(b * NNV + i0 + m) * CC + c0 + kkA];
            const int kkB = ((r & 3) * 2) + ((r >> 2) * 16) + 8 * hi;   // B layout
            const float* wp = &w[o * CC + c0 + kkB];
            bm.u[r] = packbf(wp[0], wp[1]);
        }
        acc = wmma_bf16(a, bm, acc);
    }
    const float sc = g[o] * rsqrtf(vr[o] + EPSV);
    const float sh = bt[o] - mu[o] * sc;
    #pragma unroll
    for (int r = 0; r < 8; ++r) {
        const int i = i0 + r + 8 * hi;          // pixel (D row)
        const unsigned short yb = f2bf(acc[r] * sc + sh);
        if (kind == 0)      qbuf[(b * NNV + i) * C8V + o] = yb;
        else if (kind == 1) kbuf[(b * NNV + i) * C8V + o] = yb;
        else                vbuf[(b * CC + o) * NNV + i]  = yb;
    }
}

// ---------------------------------------------------------------------------
// Kernel 2: flash attention + gamma*out + x residual.
// grid = (N/16, B); 1 wave/block; 16 queries/block; j streamed in chunks of 64.
// Per-row softmax state (m,l) replicated per-lane at index r+8*hi (D layout).
// Row sums via WMMA against an all-ones B tile; row maxes via xor-butterfly
// shuffles inside each 16-lane half (matches the D-tile layout).
// ---------------------------------------------------------------------------
__global__ void __launch_bounds__(32) attn_kernel(
        const unsigned short* __restrict__ qbuf,
        const unsigned short* __restrict__ kbuf,
        const unsigned short* __restrict__ vbuf,
        const float* __restrict__ x,
        const float* __restrict__ gamma,
        float* __restrict__ out) {
    __shared__ float sS[16 * 64];   // 16 rows x 64 cols score scratch (4 KB)
    __shared__ float sM[16];        // row max, row-major broadcast for P build
    const int lane = threadIdx.x;
    const int m  = lane & 15;
    const int hi = lane >> 4;
    const int b  = blockIdx.y;
    const int i0 = blockIdx.x * 16;

    // Q A-tile (fixed for the whole block)
    Mat16 qa;
    #pragma unroll
    for (int r = 0; r < 8; ++r) {
        const int kk = ((r & 3) * 2) + ((r >> 2) * 16) + 8 * hi;
        qa.u[r] = *(const unsigned int*)&qbuf[(b * NNV + i0 + m) * C8V + kk];
    }

    Mat16 ones;                      // B tile of bf16 1.0 for row-sum WMMA
    #pragma unroll
    for (int r = 0; r < 8; ++r) ones.u[r] = 0x3F803F80u;

    const v8f zero8 = {};
    v8f acc[16];
    #pragma unroll
    for (int ct = 0; ct < 16; ++ct) acc[ct] = zero8;
    float mr[8], lr[8];              // running max / sum for row r+8*hi
    #pragma unroll
    for (int r = 0; r < 8; ++r) { mr[r] = -3.0e38f; lr[r] = 0.0f; }

    for (int j0 = 0; j0 < NNV; j0 += 64) {
        // ---- S = Q K^T for 64 key columns (4 WMMAs) ----
        v8f s[4];
        #pragma unroll
        for (int t = 0; t < 4; ++t) {
            Mat16 kt;
            #pragma unroll
            for (int r = 0; r < 8; ++r) {
                const int kk = ((r & 3) * 2) + ((r >> 2) * 16) + 8 * hi; // B layout, K = channel
                kt.u[r] = *(const unsigned int*)
                    &kbuf[(b * NNV + j0 + t * 16 + m) * C8V + kk];
            }
            s[t] = wmma_bf16(qa, kt, zero8);
        }
        // ---- stash S for the layout change D -> A ----
        #pragma unroll
        for (int t = 0; t < 4; ++t)
            #pragma unroll
            for (int r = 0; r < 8; ++r)
                sS[(r + 8 * hi) * 64 + t * 16 + m] = s[t][r];
        // ---- row max: elementwise over tiles, then xor-butterfly in-half ----
        float rmx[8];
        #pragma unroll
        for (int r = 0; r < 8; ++r)
            rmx[r] = fmaxf(fmaxf(s[0][r], s[1][r]), fmaxf(s[2][r], s[3][r]));
        #pragma unroll
        for (int mask = 1; mask <= 8; mask <<= 1)
            #pragma unroll
            for (int r = 0; r < 8; ++r)
                rmx[r] = fmaxf(rmx[r], __shfl_xor(rmx[r], mask));
        v8f cfv;
        #pragma unroll
        for (int r = 0; r < 8; ++r) {
            const float mn = fmaxf(mr[r], rmx[r]);
            cfv[r] = __expf(mr[r] - mn);                     // correction factor
            mr[r] = mn;
        }
        if (m == 0) {                                        // lanes 0 & 16: 8 rows each
            #pragma unroll
            for (int r = 0; r < 8; ++r) sM[r + 8 * hi] = mr[r];
        }
        const float mrow = sM[m];                            // row-major broadcast
        // ---- P tiles (A-matrix layout, bf16), one exp per score ----
        Mat16 pa0, pa1;
        #pragma unroll
        for (int r = 0; r < 8; ++r) {
            const int kk = ((r & 3) * 2) + ((r >> 2) * 16) + 8 * hi;
            pa0.u[r] = packbf(__expf(sS[m * 64 + kk]          - mrow),
                              __expf(sS[m * 64 + kk + 1]      - mrow));
            pa1.u[r] = packbf(__expf(sS[m * 64 + 32 + kk]     - mrow),
                              __expf(sS[m * 64 + 32 + kk + 1] - mrow));
        }
        // ---- row sums = P x ones (2 WMMAs), lands at index r+8*hi per lane ----
        v8f rs = wmma_bf16(pa0, ones, zero8);
        rs = wmma_bf16(pa1, ones, rs);
        #pragma unroll
        for (int r = 0; r < 8; ++r) lr[r] = lr[r] * cfv[r] + rs[r];
        // ---- rescale running O (vector mul -> packed f32 pairs) ----
        #pragma unroll
        for (int ct = 0; ct < 16; ++ct) acc[ct] *= cfv;
        // ---- O += P V  (32 WMMAs over 256 channels) ----
        #pragma unroll
        for (int ct = 0; ct < 16; ++ct) {
            Mat16 vt0, vt1;
            const int vbase = (b * CC + ct * 16 + m) * NNV + j0;
            #pragma unroll
            for (int r = 0; r < 8; ++r) {
                const int kk = ((r & 3) * 2) + ((r >> 2) * 16) + 8 * hi; // B layout, K = j
                vt0.u[r] = *(const unsigned int*)&vbuf[vbase + kk];
                vt1.u[r] = *(const unsigned int*)&vbuf[vbase + 32 + kk];
            }
            acc[ct] = wmma_bf16(pa0, vt0, acc[ct]);
            acc[ct] = wmma_bf16(pa1, vt1, acc[ct]);
        }
    }

    // ---- epilogue: /rowsum, *gamma, +x residual ----
    const float gsc = gamma[0];
    float linv[8];
    #pragma unroll
    for (int r = 0; r < 8; ++r) linv[r] = 1.0f / lr[r];
    #pragma unroll
    for (int ct = 0; ct < 16; ++ct) {
        const int c = ct * 16 + m;
        #pragma unroll
        for (int r = 0; r < 8; ++r) {
            const int i   = i0 + r + 8 * hi;
            const int idx = (b * CC + c) * NNV + i;
            out[idx] = gsc * acc[ct][r] * linv[r] + x[idx];
        }
    }
}

// ---------------------------------------------------------------------------
extern "C" void kernel_launch(void* const* d_in, const int* in_sizes, int n_in,
                              void* d_out, int out_size, void* d_ws, size_t ws_size,
                              hipStream_t stream) {
    const float* x     = (const float*)d_in[0];
    const float* wq    = (const float*)d_in[1];
    const float* bnq_g = (const float*)d_in[2];
    const float* bnq_b = (const float*)d_in[3];
    const float* bnq_m = (const float*)d_in[4];
    const float* bnq_v = (const float*)d_in[5];
    const float* wk    = (const float*)d_in[6];
    const float* bnk_g = (const float*)d_in[7];
    const float* bnk_b = (const float*)d_in[8];
    const float* bnk_m = (const float*)d_in[9];
    const float* bnk_v = (const float*)d_in[10];
    const float* wv    = (const float*)d_in[11];
    const float* bnv_g = (const float*)d_in[12];
    const float* bnv_b = (const float*)d_in[13];
    const float* bnv_m = (const float*)d_in[14];
    const float* bnv_v = (const float*)d_in[15];
    const float* gamma = (const float*)d_in[16];
    float* out = (float*)d_out;

    char* ws = (char*)d_ws;
    unsigned short* xT   = (unsigned short*)ws;                          // 8 MB
    unsigned short* qbuf = (unsigned short*)(ws + (size_t)BB*NNV*CC*2);  // 1 MB
    unsigned short* kbuf = qbuf + (size_t)BB * NNV * C8V;                // 1 MB
    unsigned short* vbuf = kbuf + (size_t)BB * NNV * C8V;                // 8 MB

    dim3 gT(NNV / 32, CC / 32, BB);
    xpose_kernel<<<gT, dim3(32, 8), 0, stream>>>(x, xT);

    dim3 gP(20, NNV / 16, BB);
    proj_kernel<<<gP, dim3(32), 0, stream>>>(xT,
        wq, bnq_g, bnq_b, bnq_m, bnq_v,
        wk, bnk_g, bnk_b, bnk_m, bnk_v,
        wv, bnv_g, bnv_b, bnv_m, bnv_v,
        qbuf, kbuf, vbuf);

    dim3 gA(NNV / 16, BB);
    attn_kernel<<<gA, dim3(32), 0, stream>>>(qbuf, kbuf, vbuf, x, gamma, out);
}
